// Gemma4MoEDecoderLayer_53815940219214
// MI455X (gfx1250) — matrix operand order repaired and checked by reading, not compiled, not run on hardware
//
#include <hip/hip_runtime.h>
#include <hip/hip_bf16.h>
#include <cstddef>

// ---------------- problem constants ----------------
#define S_   1024
#define D_   2048
#define H_   8
#define KV_  4
#define DH_  256
#define DFF_ 8192
#define E_   8
#define DFE_ 1024
#define EPS_ 1e-6f
#define KS_  64   // K elements per LDS stage

// ---------------- types ----------------
typedef __attribute__((ext_vector_type(16))) __bf16        v16bf;
typedef __attribute__((ext_vector_type(8)))  float         v8f;
typedef __attribute__((ext_vector_type(4)))  unsigned int  u32x4;
typedef __attribute__((ext_vector_type(8)))  int           i32x8;
typedef __attribute__((ext_vector_type(4)))  int           i32x4;
struct U32x8 { u32x4 lo, hi; };

// ---------------- small helpers ----------------
__device__ __forceinline__ unsigned short f2bf(float f) {
  unsigned int u = __builtin_bit_cast(unsigned int, f);
  u += 0x7FFFu + ((u >> 16) & 1u);   // round-to-nearest-even
  return (unsigned short)(u >> 16);
}

__device__ __forceinline__ float gelu_tanh(float x) {
  float x3 = x * x * x;
  return 0.5f * x * (1.f + tanhf(0.7978845608028654f * (x + 0.044715f * x3)));
}

__device__ __forceinline__ float blockReduceSum(float v) {
  __shared__ float red[9];
  __syncthreads();
#pragma unroll
  for (int off = 16; off > 0; off >>= 1) v += __shfl_down(v, off, 32);
  const int lane = threadIdx.x & 31, w = threadIdx.x >> 5;
  if (lane == 0) red[w] = v;
  __syncthreads();
  if (threadIdx.x == 0) {
    float s = 0.f;
    const int nw = (blockDim.x + 31) >> 5;
    for (int i = 0; i < nw; ++i) s += red[i];
    red[8] = s;
  }
  __syncthreads();
  return red[8];
}

__device__ __forceinline__ float blockReduceMax(float v) {
  __shared__ float red[9];
  __syncthreads();
#pragma unroll
  for (int off = 16; off > 0; off >>= 1) v = fmaxf(v, __shfl_down(v, off, 32));
  const int lane = threadIdx.x & 31, w = threadIdx.x >> 5;
  if (lane == 0) red[w] = v;
  __syncthreads();
  if (threadIdx.x == 0) {
    float s = -3.4e38f;
    const int nw = (blockDim.x + 31) >> 5;
    for (int i = 0; i < nw; ++i) s = fmaxf(s, red[i]);
    red[8] = s;
  }
  __syncthreads();
  return red[8];
}

// ---------------- WMMA fragment loads (ISA lane layouts) ----------------
// A (16x32 bf16, row-major, stride ld): lane<16 -> row=lane, K in {0..7,16..23};
// lane>=16 -> row=lane-16, K in {8..15,24..31}.
__device__ __forceinline__ v16bf load_frag_a(const unsigned short* __restrict__ p,
                                             int ld, int lane) {
  const int r = lane & 15, hi = lane >> 4;
  const unsigned short* base = p + (size_t)r * ld + hi * 8;
  U32x8 u;
  u.lo = *(const u32x4*)(base);        // K = hi*8 .. hi*8+7
  u.hi = *(const u32x4*)(base + 16);   // K = 16+hi*8 .. 16+hi*8+7
  return __builtin_bit_cast(v16bf, u);
}
// B stored transposed [N,K] row-major (stride ld): lane holds col n=lane&15,
// K in {(lane>>4)*8..+7, 16+(lane>>4)*8..+7} (same order as A).
__device__ __forceinline__ v16bf load_frag_b(const unsigned short* __restrict__ p,
                                             int ld, int lane) {
  const int n = lane & 15, hi = lane >> 4;
  const unsigned short* base = p + (size_t)n * ld + hi * 8;
  U32x8 u;
  u.lo = *(const u32x4*)(base);
  u.hi = *(const u32x4*)(base + 16);
  return __builtin_bit_cast(v16bf, u);
}

__device__ __forceinline__ v8f wmma_bf16(v16bf a, v16bf b, v8f c) {
  return __builtin_amdgcn_wmma_f32_16x16x32_bf16(false, a, false, b,
                                                 (short)0, c, false, false);
}

// ---------------- Tensor Data Mover: 2D tile global->LDS ----------------
// D# per CDNA5 ISA 8.3/8.4. group0 = {count=1,valid | lds_addr | global_addr | type=2}
// group1 = {data_size=2B, tensor_dim0/1, tile_dim0=cols, tile_dim1=rows, dim0_stride}
__device__ __forceinline__ void tdm_load_2d(unsigned lds_off,
                                            const unsigned short* gptr,
                                            int rows, int cols, int stride_elems) {
  unsigned long long ga = (unsigned long long)(size_t)gptr;
  u32x4 g0;
  g0[0] = 1u;                                   // count=1 (valid), user mode
  g0[1] = lds_off;                              // LDS byte address
  g0[2] = (unsigned)(ga & 0xFFFFFFFFu);         // global_addr[31:0]
  g0[3] = (unsigned)((ga >> 32) & 0x01FFFFFFu)  // global_addr[56:32]
        | 0x80000000u;                          // type=2 ("image")
  const unsigned td0 = (unsigned)cols, td1 = (unsigned)rows;
  const unsigned long long st = (unsigned long long)(unsigned)stride_elems;
  i32x8 g1;
  g1[0] = (int)(1u << 16);                                        // data_size: 2 bytes
  g1[1] = (int)((td0 & 0xFFFFu) << 16);                           // tensor_dim0 lo
  g1[2] = (int)((td0 >> 16) | ((td1 & 0xFFFFu) << 16));           // dim0 hi | dim1 lo
  g1[3] = (int)((td1 >> 16) | ((td0 & 0xFFFFu) << 16));           // dim1 hi | tile_dim0
  g1[4] = (int)(td1 & 0xFFFFu);                                   // tile_dim1 (tile_dim2=0)
  g1[5] = (int)(unsigned)(st & 0xFFFFFFFFull);                    // dim0_stride lo
  g1[6] = (int)(unsigned)((st >> 32) & 0xFFFFull);                // dim0_stride hi
  g1[7] = 0;
  i32x4 z4 = {0, 0, 0, 0};
#if defined(__clang_major__) && (__clang_major__ >= 23)
  i32x8 z8 = {0, 0, 0, 0, 0, 0, 0, 0};
  __builtin_amdgcn_tensor_load_to_lds(g0, g1, z4, z4, z8, 0);
#else
  __builtin_amdgcn_tensor_load_to_lds(g0, g1, z4, z4, 0);
#endif
}

// C/D layout: VGPR j -> lanes 0-15: M=j, N=lane; lanes 16-31: M=j+8, N=lane-16
__device__ __forceinline__ void store_tile(float* __restrict__ C, int ldc,
                                           int mBase, int nBase, v8f acc, int lane,
                                           const float* __restrict__ rowScale,
                                           int rss, int accumulate) {
  const int col = nBase + (lane & 15);
  const int rb  = mBase + (lane >> 4) * 8;
#pragma unroll
  for (int j = 0; j < 8; ++j) {
    const int row = rb + j;
    float v = acc[j];
    if (rowScale) v *= rowScale[(size_t)row * rss];
    float* p = C + (size_t)row * ldc + col;
    *p = accumulate ? (*p + v) : v;
  }
}

// ---------------- GEMM: C[M,N] (+)= rowScale .* (A[M,K] @ Bt[N,K]^T) ----------------
// Requires M%64==0, N%128==0, K%KS_==0 (true for every GEMM in this layer).
// TDM double-buffered LDS staging: transfer of stage k+1 overlaps WMMA of stage k.
__global__ __launch_bounds__(256) void gemm_bf16_nt(
    const unsigned short* __restrict__ A, int lda,
    const unsigned short* __restrict__ B, int ldb,
    float* __restrict__ C, int ldc,
    int M, int N, int K,
    const float* __restrict__ rowScale, int rss, int accumulate) {
  __shared__ __align__(16) unsigned short s_a[2][64][KS_];
  __shared__ __align__(16) unsigned short s_b[2][128][KS_];
  const int tid  = threadIdx.x;
  const int lane = tid & 31;
  const int wave = tid >> 5;          // 8 waves: 2 (M) x 4 (N)
  const int wm = wave >> 2;
  const int wn = wave & 3;
  const int mBlk = blockIdx.y * 64;
  const int nBlk = blockIdx.x * 128;

  v8f acc00{}, acc01{}, acc10{}, acc11{};
  const int nk = K / KS_;

  auto issue = [&](int kt) {
    const int k0 = kt * KS_;
    const int buf = kt & 1;
    tdm_load_2d((unsigned)(size_t)(void*)&s_a[buf][0][0],
                A + (size_t)mBlk * lda + k0, 64, KS_, lda);
    tdm_load_2d((unsigned)(size_t)(void*)&s_b[buf][0][0],
                B + (size_t)nBlk * ldb + k0, 128, KS_, ldb);
  };

  if (wave == 0) issue(0);
  for (int kt = 0; kt < nk; ++kt) {
    if (wave == 0) __builtin_amdgcn_s_wait_tensorcnt(0);  // stage kt landed in LDS
    __syncthreads();
    if (wave == 0 && kt + 1 < nk) issue(kt + 1);          // overlap next transfer
    const int buf = kt & 1;
    const unsigned short* sa = &s_a[buf][0][0];
    const unsigned short* sb = &s_b[buf][0][0];
#pragma unroll
    for (int kk = 0; kk < KS_; kk += 32) {
      v16bf a0 = load_frag_a(sa + (wm * 32 +  0) * KS_ + kk, KS_, lane);
      v16bf a1 = load_frag_a(sa + (wm * 32 + 16) * KS_ + kk, KS_, lane);
      v16bf b0 = load_frag_b(sb + (wn * 32 +  0) * KS_ + kk, KS_, lane);
      v16bf b1 = load_frag_b(sb + (wn * 32 + 16) * KS_ + kk, KS_, lane);
      acc00 = wmma_bf16(a0, b0, acc00);
      acc01 = wmma_bf16(a0, b1, acc01);
      acc10 = wmma_bf16(a1, b0, acc10);
      acc11 = wmma_bf16(a1, b1, acc11);
    }
    __syncthreads();   // compute done before this buffer is refilled
  }
  const int mBase = mBlk + wm * 32;
  const int nBase = nBlk + wn * 32;
  store_tile(C, ldc, mBase,      nBase,      acc00, lane, rowScale, rss, accumulate);
  store_tile(C, ldc, mBase,      nBase + 16, acc01, lane, rowScale, rss, accumulate);
  store_tile(C, ldc, mBase + 16, nBase,      acc10, lane, rowScale, rss, accumulate);
  store_tile(C, ldc, mBase + 16, nBase + 16, acc11, lane, rowScale, rss, accumulate);
}

// ---------------- transpose fp32 [R,C] (row stride inLd) -> bf16 [C,R] ----------------
__global__ __launch_bounds__(256) void transpose_to_bf16(
    const float* __restrict__ in, int inLd,
    unsigned short* __restrict__ out, int R, int C) {
  __shared__ float tile[32][33];
  const int c0 = blockIdx.x * 32, r0 = blockIdx.y * 32;
  const int tx = threadIdx.x & 31, ty = threadIdx.x >> 5;
  for (int i = ty; i < 32; i += 8) {
    int r = r0 + i, c = c0 + tx;
    tile[i][tx] = (r < R && c < C) ? in[(size_t)r * inLd + c] : 0.f;
  }
  __syncthreads();
  for (int i = ty; i < 32; i += 8) {
    int c = c0 + i, r = r0 + tx;
    if (c < C && r < R) out[(size_t)c * R + r] = f2bf(tile[tx][i]);
  }
}

// ---------------- elementwise / norm kernels ----------------
__global__ __launch_bounds__(256) void rmsnorm_to_bf16(
    const float* __restrict__ x, const float* __restrict__ w,
    unsigned short* __restrict__ out, int dim) {
  const int r = blockIdx.x;
  const float* row = x + (size_t)r * dim;
  float ss = 0.f;
  for (int d = threadIdx.x; d < dim; d += blockDim.x) { float v = row[d]; ss += v * v; }
  ss = blockReduceSum(ss);
  const float sc = rsqrtf(ss / dim + EPS_);
  for (int d = threadIdx.x; d < dim; d += blockDim.x)
    out[(size_t)r * dim + d] = f2bf(row[d] * sc * (1.f + w[d]));
}

__global__ __launch_bounds__(256) void residual_rmsnorm_add(
    const float* __restrict__ o, const float* __restrict__ w,
    const float* __restrict__ resid, float* __restrict__ x1, int dim) {
  const int r = blockIdx.x;
  const float* row = o + (size_t)r * dim;
  float ss = 0.f;
  for (int d = threadIdx.x; d < dim; d += blockDim.x) { float v = row[d]; ss += v * v; }
  ss = blockReduceSum(ss);
  const float sc = rsqrtf(ss / dim + EPS_);
  for (int d = threadIdx.x; d < dim; d += blockDim.x)
    x1[(size_t)r * dim + d] = resid[(size_t)r * dim + d] + row[d] * sc * (1.f + w[d]);
}

// per-(token,head) RMSNorm + RoPE, fp32 in -> bf16 out. layout [S, nH, DH]
__global__ __launch_bounds__(DH_) void qk_norm_rope(
    const float* __restrict__ q, const float* __restrict__ cosv,
    const float* __restrict__ sinv, const float* __restrict__ nw,
    unsigned short* __restrict__ out, int nH) {
  const int s = blockIdx.x / nH;
  const int h = blockIdx.x % nH;
  const int d = threadIdx.x;
  const float* row = q + ((size_t)s * nH + h) * DH_;
  float v = row[d];
  float ss = blockReduceSum(v * v);
  const float sc = rsqrtf(ss / DH_ + EPS_);
  const int d2 = (d < DH_ / 2) ? d + DH_ / 2 : d - DH_ / 2;
  float n  = v * sc * (1.f + nw[d]);
  float n2 = row[d2] * sc * (1.f + nw[d2]);
  float rot = (d < DH_ / 2) ? -n2 : n2;
  float o = n * cosv[(size_t)s * DH_ + d] + rot * sinv[(size_t)s * DH_ + d];
  out[((size_t)s * nH + h) * DH_ + d] = f2bf(o);
}

// causal softmax over one score row, writes bf16 probs (0 beyond diagonal)
__global__ __launch_bounds__(256) void softmax_causal(
    const float* __restrict__ scores, unsigned short* __restrict__ probs,
    float scale) {
  const int q = blockIdx.x;
  const int h = blockIdx.y;
  const float* row = scores + ((size_t)h * S_ + q) * (size_t)S_;
  unsigned short* prow = probs + ((size_t)h * S_ + q) * (size_t)S_;
  const int L = q + 1;
  float m = -3.4e38f;
  for (int k = threadIdx.x; k < L; k += blockDim.x) m = fmaxf(m, row[k] * scale);
  m = blockReduceMax(m);
  float sum = 0.f;
  for (int k = threadIdx.x; k < L; k += blockDim.x) sum += __expf(row[k] * scale - m);
  sum = blockReduceSum(sum);
  const float inv = 1.f / sum;
  for (int k = threadIdx.x; k < S_; k += blockDim.x) {
    float p = (k < L) ? __expf(row[k] * scale - m) * inv : 0.f;
    prow[k] = f2bf(p);
  }
}

__global__ __launch_bounds__(256) void glu_to_bf16(
    const float* __restrict__ g, const float* __restrict__ u,
    unsigned short* __restrict__ out, size_t n) {
  size_t i = (size_t)blockIdx.x * blockDim.x + threadIdx.x;
  if (i < n) out[i] = f2bf(gelu_tanh(g[i]) * u[i]);
}

__global__ __launch_bounds__(256) void to_bf16(
    const float* __restrict__ in, unsigned short* __restrict__ out, size_t n) {
  size_t i = (size_t)blockIdx.x * blockDim.x + threadIdx.x;
  if (i < n) out[i] = f2bf(in[i]);
}

__global__ __launch_bounds__(256) void zero_f32(float* __restrict__ p, size_t n) {
  size_t i = (size_t)blockIdx.x * blockDim.x + threadIdx.x;
  if (i < n) p[i] = 0.f;
}

// router: combine[t,e] = normalized top-2 softmax weights
__global__ __launch_bounds__(256) void routing_kernel(
    const float* __restrict__ x1, const float* __restrict__ gsc,
    const float* __restrict__ gw, float* __restrict__ combine) {
  const int t = blockIdx.x;
  const float* row = x1 + (size_t)t * D_;
  float ss = 0.f;
  for (int d = threadIdx.x; d < D_; d += blockDim.x) { float v = row[d]; ss += v * v; }
  ss = blockReduceSum(ss);
  const float coef = rsqrtf(ss / D_ + EPS_) * rsqrtf((float)D_);
  float le[E_];
#pragma unroll
  for (int e = 0; e < E_; ++e) le[e] = 0.f;
  for (int d = threadIdx.x; d < D_; d += blockDim.x) {
    float gn = row[d] * coef * gsc[d];
#pragma unroll
    for (int e = 0; e < E_; ++e) le[e] += gn * gw[(size_t)d * E_ + e];
  }
  float logit[E_];
#pragma unroll
  for (int e = 0; e < E_; ++e) logit[e] = blockReduceSum(le[e]);
  if (threadIdx.x == 0) {
    float m = -3.4e38f;
#pragma unroll
    for (int e = 0; e < E_; ++e) m = fmaxf(m, logit[e]);
    float p[E_], s = 0.f;
#pragma unroll
    for (int e = 0; e < E_; ++e) { p[e] = __expf(logit[e] - m); s += p[e]; }
#pragma unroll
    for (int e = 0; e < E_; ++e) p[e] /= s;
    int i1 = 0;
#pragma unroll
    for (int e = 1; e < E_; ++e) if (p[e] > p[i1]) i1 = e;
    int i2 = (i1 == 0) ? 1 : 0;
#pragma unroll
    for (int e = 0; e < E_; ++e) if (e != i1 && p[e] > p[i2]) i2 = e;
    float wsum = fmaxf(p[i1] + p[i2], 1e-20f);
#pragma unroll
    for (int e = 0; e < E_; ++e)
      combine[(size_t)t * E_ + e] =
          (e == i1) ? p[i1] / wsum : ((e == i2) ? p[i2] / wsum : 0.f);
  }
}

// out = (x1 + rmsnorm(rmsnorm(dmlp,w1) + rmsnorm(y,w2), wf)) * layer_scalar
__global__ __launch_bounds__(256) void final_fuse(
    const float* __restrict__ x1, const float* __restrict__ dmlp,
    const float* __restrict__ y, const float* __restrict__ w1,
    const float* __restrict__ w2, const float* __restrict__ wf,
    const float* __restrict__ lsp, float* __restrict__ out) {
  __shared__ float srow[D_];
  const int r = blockIdx.x;
  const float* dr = dmlp + (size_t)r * D_;
  const float* yr = y + (size_t)r * D_;
  float ssd = 0.f, ssy = 0.f;
  for (int d = threadIdx.x; d < D_; d += blockDim.x) {
    float a = dr[d], b = yr[d];
    ssd += a * a; ssy += b * b;
  }
  ssd = blockReduceSum(ssd);
  ssy = blockReduceSum(ssy);
  const float sd = rsqrtf(ssd / D_ + EPS_);
  const float sy = rsqrtf(ssy / D_ + EPS_);
  float sss = 0.f;
  for (int d = threadIdx.x; d < D_; d += blockDim.x) {
    float v = dr[d] * sd * (1.f + w1[d]) + yr[d] * sy * (1.f + w2[d]);
    srow[d] = v;
    sss += v * v;
  }
  sss = blockReduceSum(sss);
  const float sc = rsqrtf(sss / D_ + EPS_);
  const float ls = lsp[0];
  for (int d = threadIdx.x; d < D_; d += blockDim.x)
    out[(size_t)r * D_ + d] =
        (x1[(size_t)r * D_ + d] + srow[d] * sc * (1.f + wf[d])) * ls;
}

// ---------------- host orchestration ----------------
extern "C" void kernel_launch(void* const* d_in, const int* in_sizes, int n_in,
                              void* d_out, int out_size, void* d_ws, size_t ws_size,
                              hipStream_t stream) {
  (void)in_sizes; (void)n_in; (void)out_size; (void)ws_size;
  const float* x          = (const float*)d_in[0];
  const float* cosv       = (const float*)d_in[1];
  const float* sinv       = (const float*)d_in[2];
  const float* wq         = (const float*)d_in[3];
  const float* wk         = (const float*)d_in[4];
  const float* wv         = (const float*)d_in[5];
  const float* wo         = (const float*)d_in[6];
  const float* q_norm_w   = (const float*)d_in[7];
  const float* k_norm_w   = (const float*)d_in[8];
  const float* ln_in      = (const float*)d_in[9];
  const float* ln_post_attn = (const float*)d_in[10];
  const float* ln_pre_ff  = (const float*)d_in[11];
  const float* ln_post_ff = (const float*)d_in[12];
  const float* ln_post_ff1 = (const float*)d_in[13];
  const float* ln_post_ff2 = (const float*)d_in[14];
  const float* ln_pre_ff2 = (const float*)d_in[15];
  const float* mlp_gate   = (const float*)d_in[16];
  const float* mlp_up     = (const float*)d_in[17];
  const float* mlp_down   = (const float*)d_in[18];
  const float* gate_scale = (const float*)d_in[19];
  const float* gate_w     = (const float*)d_in[20];
  const float* eg         = (const float*)d_in[21];
  const float* eu         = (const float*)d_in[22];
  const float* ed         = (const float*)d_in[23];
  const float* layer_scalar = (const float*)d_in[24];
  float* out = (float*)d_out;

  // workspace bump allocator (256B aligned)
  char* base = (char*)d_ws;
  size_t off = 0;
  auto alloc = [&](size_t bytes) -> void* {
    off = (off + 255) & ~(size_t)255;
    void* p = base + off;
    off += bytes;
    return p;
  };
  typedef unsigned short bfp;
  bfp* wq_t   = (bfp*)alloc((size_t)(H_*DH_) * D_ * 2);
  bfp* wk_t   = (bfp*)alloc((size_t)(KV_*DH_) * D_ * 2);
  bfp* wv_t   = (bfp*)alloc((size_t)(KV_*DH_) * D_ * 2);
  bfp* wo_t   = (bfp*)alloc((size_t)D_ * (H_*DH_) * 2);
  bfp* gate_t = (bfp*)alloc((size_t)DFF_ * D_ * 2);
  bfp* up_t   = (bfp*)alloc((size_t)DFF_ * D_ * 2);
  bfp* down_t = (bfp*)alloc((size_t)D_ * DFF_ * 2);
  bfp* egt    = (bfp*)alloc((size_t)DFE_ * D_ * 2);
  bfp* eut    = (bfp*)alloc((size_t)DFE_ * D_ * 2);
  bfp* edt    = (bfp*)alloc((size_t)D_ * DFE_ * 2);
  bfp* h_bf   = (bfp*)alloc((size_t)S_ * D_ * 2);
  float* qf   = (float*)alloc((size_t)S_ * H_ * DH_ * 4);
  float* kf   = (float*)alloc((size_t)S_ * KV_ * DH_ * 4);
  float* vf   = (float*)alloc((size_t)S_ * KV_ * DH_ * 4);
  bfp* q_bf   = (bfp*)alloc((size_t)S_ * H_ * DH_ * 2);
  bfp* k_bf   = (bfp*)alloc((size_t)S_ * KV_ * DH_ * 2);
  bfp* vt_bf  = (bfp*)alloc((size_t)KV_ * DH_ * S_ * 2);
  float* scores = (float*)alloc((size_t)H_ * S_ * S_ * 4);
  bfp* probs  = (bfp*)alloc((size_t)H_ * S_ * S_ * 2);
  float* of   = (float*)alloc((size_t)S_ * H_ * DH_ * 4);
  bfp* o_bf   = (bfp*)alloc((size_t)S_ * H_ * DH_ * 2);
  float* aproj = (float*)alloc((size_t)S_ * D_ * 4);
  float* x1   = (float*)alloc((size_t)S_ * D_ * 4);
  bfp* ffin_bf = (bfp*)alloc((size_t)S_ * D_ * 2);
  float* gatef = (float*)alloc((size_t)S_ * DFF_ * 4);
  float* upf  = (float*)alloc((size_t)S_ * DFF_ * 4);
  bfp* act_bf = (bfp*)alloc((size_t)S_ * DFF_ * 2);
  float* dmlp = (float*)alloc((size_t)S_ * D_ * 4);
  bfp* t_bf   = (bfp*)alloc((size_t)S_ * D_ * 2);
  float* combine = (float*)alloc((size_t)S_ * E_ * 4);
  float* egout = (float*)alloc((size_t)S_ * DFE_ * 4);
  float* euout = (float*)alloc((size_t)S_ * DFE_ * 4);
  bfp* eact_bf = (bfp*)alloc((size_t)S_ * DFE_ * 2);
  float* y    = (float*)alloc((size_t)S_ * D_ * 4);

  auto tr = [&](const float* in, int inLd, bfp* o, int R, int C) {
    dim3 g((C + 31) / 32, (R + 31) / 32);
    transpose_to_bf16<<<g, 256, 0, stream>>>(in, inLd, o, R, C);
  };
  auto gemm = [&](const bfp* A, int lda, const bfp* B, int ldb,
                  float* C, int ldc, int M, int N, int K,
                  const float* rs, int rss, int accum) {
    dim3 g(N / 128, M / 64);
    gemm_bf16_nt<<<g, 256, 0, stream>>>(A, lda, B, ldb, C, ldc, M, N, K, rs, rss, accum);
  };

  // 1) weight conversion to bf16, transposed to [N,K]
  tr(wq, H_*DH_, wq_t, D_, H_*DH_);
  tr(wk, KV_*DH_, wk_t, D_, KV_*DH_);
  tr(wv, KV_*DH_, wv_t, D_, KV_*DH_);
  tr(wo, D_, wo_t, H_*DH_, D_);
  tr(mlp_gate, DFF_, gate_t, D_, DFF_);
  tr(mlp_up,   DFF_, up_t,   D_, DFF_);
  tr(mlp_down, D_,   down_t, DFF_, D_);

  // 2) input norm -> bf16, QKV projections
  rmsnorm_to_bf16<<<S_, 256, 0, stream>>>(x, ln_in, h_bf, D_);
  gemm(h_bf, D_, wq_t, D_, qf, H_*DH_, S_, H_*DH_, D_, nullptr, 0, 0);
  gemm(h_bf, D_, wk_t, D_, kf, KV_*DH_, S_, KV_*DH_, D_, nullptr, 0, 0);
  gemm(h_bf, D_, wv_t, D_, vf, KV_*DH_, S_, KV_*DH_, D_, nullptr, 0, 0);

  // 3) per-head QK norm + RoPE -> bf16; V transposed per KV head
  qk_norm_rope<<<S_*H_, DH_, 0, stream>>>(qf, cosv, sinv, q_norm_w, q_bf, H_);
  qk_norm_rope<<<S_*KV_, DH_, 0, stream>>>(kf, cosv, sinv, k_norm_w, k_bf, KV_);
  for (int kv = 0; kv < KV_; ++kv)
    tr(vf + (size_t)kv * DH_, KV_*DH_, vt_bf + (size_t)kv * DH_ * S_, S_, DH_);

  // 4) attention: scores -> causal softmax -> P@V
  for (int h = 0; h < H_; ++h)
    gemm(q_bf + (size_t)h * DH_, H_*DH_, k_bf + (size_t)(h/2) * DH_, KV_*DH_,
         scores + (size_t)h * S_ * S_, S_, S_, S_, DH_, nullptr, 0, 0);
  {
    dim3 g(S_, H_);
    softmax_causal<<<g, 256, 0, stream>>>(scores, probs, rsqrtf((float)DH_));
  }
  for (int h = 0; h < H_; ++h)
    gemm(probs + (size_t)h * S_ * S_, S_, vt_bf + (size_t)(h/2) * DH_ * S_, S_,
         of + (size_t)h * DH_, H_*DH_, S_, DH_, S_, nullptr, 0, 0);

  // 5) output projection + post-attn norm + residual
  to_bf16<<<(S_*H_*DH_ + 255) / 256, 256, 0, stream>>>(of, o_bf, (size_t)S_*H_*DH_);
  gemm(o_bf, H_*DH_, wo_t, H_*DH_, aproj, D_, S_, D_, H_*DH_, nullptr, 0, 0);
  residual_rmsnorm_add<<<S_, 256, 0, stream>>>(aproj, ln_post_attn, x, x1, D_);

  // 6) dense MLP
  rmsnorm_to_bf16<<<S_, 256, 0, stream>>>(x1, ln_pre_ff, ffin_bf, D_);
  gemm(ffin_bf, D_, gate_t, D_, gatef, DFF_, S_, DFF_, D_, nullptr, 0, 0);
  gemm(ffin_bf, D_, up_t,   D_, upf,   DFF_, S_, DFF_, D_, nullptr, 0, 0);
  glu_to_bf16<<<(S_*DFF_ + 255) / 256, 256, 0, stream>>>(gatef, upf, act_bf, (size_t)S_*DFF_);
  gemm(act_bf, DFF_, down_t, DFF_, dmlp, D_, S_, D_, DFF_, nullptr, 0, 0);

  // 7) MoE: routing + experts (combine fused into down-proj epilogue)
  rmsnorm_to_bf16<<<S_, 256, 0, stream>>>(x1, ln_pre_ff2, t_bf, D_);
  routing_kernel<<<S_, 256, 0, stream>>>(x1, gate_scale, gate_w, combine);
  zero_f32<<<(S_*D_ + 255) / 256, 256, 0, stream>>>(y, (size_t)S_*D_);
  for (int e = 0; e < E_; ++e) {
    tr(eg + (size_t)e * D_ * DFE_, DFE_, egt, D_, DFE_);
    tr(eu + (size_t)e * D_ * DFE_, DFE_, eut, D_, DFE_);
    tr(ed + (size_t)e * DFE_ * D_, D_,   edt, DFE_, D_);
    gemm(t_bf, D_, egt, D_, egout, DFE_, S_, DFE_, D_, nullptr, 0, 0);
    gemm(t_bf, D_, eut, D_, euout, DFE_, S_, DFE_, D_, nullptr, 0, 0);
    glu_to_bf16<<<(S_*DFE_ + 255) / 256, 256, 0, stream>>>(egout, euout, eact_bf, (size_t)S_*DFE_);
    gemm(eact_bf, DFE_, edt, DFE_, y, D_, S_, D_, DFE_, combine + e, E_, 1);
  }

  // 8) final triple-norm fusion + residual + layer scalar
  final_fuse<<<S_, 256, 0, stream>>>(x1, dmlp, y, ln_post_ff1, ln_post_ff2,
                                     ln_post_ff, layer_scalar, out);
}
